// AttentionBlock_171798692360
// MI455X (gfx1250) — compile-verified, matrix operand order checked by reading, not run on hardware
//
#include <hip/hip_runtime.h>

typedef __attribute__((ext_vector_type(8)))  __bf16 v8bf;
typedef __attribute__((ext_vector_type(16))) __bf16 v16bf;
typedef __attribute__((ext_vector_type(8)))  float  v8f;
typedef __attribute__((ext_vector_type(4)))  unsigned int u32x4;
typedef __attribute__((ext_vector_type(8)))  int i32x8;
typedef __attribute__((ext_vector_type(4)))  int i32x4;

#define HW 4096
#define CCH 256
#define NH 4
#define DH 64

// ---------------------------------------------------------------------------
// WMMA helpers (CDNA5 wave32, 16x16x32 bf16 -> f32)
// A fragment (16x32 bf16, row-major [M][K]): lane m = lane&15, kh = (lane>>4)*8,
//   regs 0..3 hold K = kh..kh+7, regs 4..7 hold K = 16+kh..16+kh+7.
// B fragment (32x16 bf16) read from memory stored as B^T = [N][K]: same pattern
//   with n = lane&15. C/D: m = r + 8*(lane>>4), n = lane&15.
// ---------------------------------------------------------------------------
__device__ __forceinline__ v16bf load_frag(const __bf16* base, int ld, int row, int lane) {
  const int m  = lane & 15;
  const int kh = (lane >> 4) << 3;
  const __bf16* p = base + (size_t)(row + m) * ld + kh;
  v8bf lo = *(const v8bf*)(p);
  v8bf hi = *(const v8bf*)(p + 16);
  return __builtin_shufflevector(lo, hi, 0,1,2,3,4,5,6,7,8,9,10,11,12,13,14,15);
}

__device__ __forceinline__ v8f wmma_bf16(v16bf a, v16bf b, v8f c) {
  return __builtin_amdgcn_wmma_f32_16x16x32_bf16(false, a, false, b, (short)0, c, false, false);
}

__device__ __forceinline__ float rmax16(float v) {
#pragma unroll
  for (int m = 1; m < 16; m <<= 1) v = fmaxf(v, __shfl_xor(v, m, 32));
  return v;
}
__device__ __forceinline__ float rsum16(float v) {
#pragma unroll
  for (int m = 1; m < 16; m <<= 1) v += __shfl_xor(v, m, 32);
  return v;
}

// ---------------------------------------------------------------------------
// Tensor Data Mover: DMA a 2D strided tile (tw_dw DWORDs x th rows, row stride
// stride_dw DWORDs) from global into LDS at lds byte offset, inserting
// pad_amount(code) DWORDs after every pad_interval(code) DWORDs (LDS pitch =
// 2^(pad_i+1) + pad_a+1 DWORDs).  D# per CDNA5 ISA ch.8: group0 = count/lds/
// global/type=2, group1 = data_size=4B, pad ctl, tensor dims, tile dims,
// dim0 stride.  Groups 2/3 zero (2D).  Tracked with TENSORcnt.
// ---------------------------------------------------------------------------
__device__ __forceinline__ unsigned lds_off(const void* p) {
  return (unsigned)(unsigned long long)(size_t)p;
}

__device__ __forceinline__ void tdm_load_2d(unsigned lds_addr, const void* gptr,
                                            unsigned tw_dw, unsigned th,
                                            unsigned stride_dw,
                                            unsigned pad_i, unsigned pad_a) {
  unsigned long long ga = (unsigned long long)(size_t)gptr;
  u32x4 g0 = { 1u,                                   // count=1, user descriptor
               lds_addr,                             // lds_addr
               (unsigned)ga,                         // global_addr[31:0]
               (unsigned)((ga >> 32) & 0x01FFFFFFu) | (2u << 30) };  // [56:32] | type=2
  i32x8 g1 = {
    (int)((2u << 16) | (1u << 20) | (pad_i << 22) | (pad_a << 25)), // data_size=4B, pad_enable
    (int)((tw_dw & 0xffffu) << 16),                                 // tensor_dim0 lo
    (int)(((tw_dw >> 16) & 0xffffu) | ((th & 0xffffu) << 16)),      // dim0 hi | tensor_dim1 lo
    (int)(((th >> 16) & 0xffffu) | ((tw_dw & 0xffffu) << 16)),      // dim1 hi | tile_dim0
    (int)(th & 0xffffu),                                            // tile_dim1 (tile_dim2=0)
    (int)stride_dw, 0, 0 };                                         // tensor_dim0_stride
  i32x4 gz = { 0, 0, 0, 0 };
#if __clang_major__ >= 23
  i32x8 gz8 = { 0, 0, 0, 0, 0, 0, 0, 0 };
  __builtin_amdgcn_tensor_load_to_lds(g0, g1, gz, gz, gz8, 0);
#else
  __builtin_amdgcn_tensor_load_to_lds(g0, g1, gz, gz, 0);
#endif
}

// ---------------------------------------------------------------------------
// Kernel 0: fp32 -> bf16 weight conversion
// ---------------------------------------------------------------------------
__global__ void __launch_bounds__(256) cvt_bf16_kernel(const float* __restrict__ in,
                                                       __bf16* __restrict__ out, int n) {
  int i = blockIdx.x * 256 + threadIdx.x;
  if (i < n) out[i] = (__bf16)in[i];
}

// ---------------------------------------------------------------------------
// Kernel 1: GroupNorm (32 groups), writes xn transposed to [b*hw, C] bf16
// ---------------------------------------------------------------------------
__global__ void __launch_bounds__(256) gn_kernel(const float* __restrict__ x,
                                                 const float* __restrict__ scale,
                                                 const float* __restrict__ bias,
                                                 __bf16* __restrict__ xn) {
  const int bb = blockIdx.x >> 5;
  const int g  = blockIdx.x & 31;
  const int tid = threadIdx.x;
  const float* xp = x + ((size_t)bb * CCH + g * 8) * HW;

  float s = 0.f, s2 = 0.f;
  for (int i = tid; i < 8 * HW; i += 256) {
    float v = xp[i];
    s += v; s2 += v * v;
  }
  __shared__ float rs[256], rq[256];
  rs[tid] = s; rq[tid] = s2;
  __syncthreads();
  for (int off = 128; off > 0; off >>= 1) {
    if (tid < off) { rs[tid] += rs[tid + off]; rq[tid] += rq[tid + off]; }
    __syncthreads();
  }
  const float mean = rs[0] * (1.f / 32768.f);
  const float var  = rq[0] * (1.f / 32768.f) - mean * mean;
  const float inv  = rsqrtf(var + 1e-5f);

  for (int i = tid; i < 8 * HW; i += 256) {
    int cl = i >> 12;
    int p  = i & (HW - 1);
    int c  = g * 8 + cl;
    float v = (xp[i] - mean) * inv * scale[c] + bias[c];
    xn[((size_t)bb * HW + p) * CCH + c] = (__bf16)v;
  }
}

// ---------------------------------------------------------------------------
// Kernel 2: QKV GEMM  out[m,n] = xn[m,:] . Wqkv[n,:] + b[n]
//   TDM double-buffered A/W tile staging; 8 waves x (32x32) WMMA sub-tiles.
// ---------------------------------------------------------------------------
__global__ void __launch_bounds__(256) qkv_gemm_kernel(const __bf16* __restrict__ A,
                                                       const __bf16* __restrict__ W,
                                                       const float* __restrict__ bias,
                                                       __bf16* __restrict__ qb,
                                                       __bf16* __restrict__ kb,
                                                       __bf16* __restrict__ vb) {
  __shared__ __align__(16) __bf16 As[2][128][40];
  __shared__ __align__(16) __bf16 Ws[2][64][40];
  const int tid = threadIdx.x, lane = tid & 31, wv = tid >> 5;
  const int wm = wv >> 1, wn = wv & 1;
  const int m0 = blockIdx.x * 128, n0 = blockIdx.y * 64;
  const bool ldr = (tid == 0);

  const v8f vzero = {0.f,0.f,0.f,0.f,0.f,0.f,0.f,0.f};
  v8f acc[2][2] = {{vzero, vzero}, {vzero, vzero}};

  if (ldr) {  // prologue DMA: tile k0=0 into buffer 0 (16 DW x rows, pitch 20 DW)
    tdm_load_2d(lds_off(&As[0][0][0]), A + (size_t)m0 * CCH, 16, 128, 128, 3, 3);
    tdm_load_2d(lds_off(&Ws[0][0][0]), W + (size_t)n0 * CCH, 16,  64, 128, 3, 3);
    __builtin_amdgcn_s_wait_tensorcnt(0);
  }
  __syncthreads();

  for (int it = 0; it < 8; ++it) {
    const int cur = it & 1, nxt = cur ^ 1;
    if (it + 1 < 8 && ldr) {  // DMA next K-slice while computing current
      const int k0 = (it + 1) * 32;
      tdm_load_2d(lds_off(&As[nxt][0][0]), A + (size_t)m0 * CCH + k0, 16, 128, 128, 3, 3);
      tdm_load_2d(lds_off(&Ws[nxt][0][0]), W + (size_t)n0 * CCH + k0, 16,  64, 128, 3, 3);
    }
    v16bf a0 = load_frag(&As[cur][0][0], 40, wm * 32,      lane);
    v16bf a1 = load_frag(&As[cur][0][0], 40, wm * 32 + 16, lane);
    v16bf b0 = load_frag(&Ws[cur][0][0], 40, wn * 32,      lane);
    v16bf b1 = load_frag(&Ws[cur][0][0], 40, wn * 32 + 16, lane);
    acc[0][0] = wmma_bf16(a0, b0, acc[0][0]);
    acc[0][1] = wmma_bf16(a0, b1, acc[0][1]);
    acc[1][0] = wmma_bf16(a1, b0, acc[1][0]);
    acc[1][1] = wmma_bf16(a1, b1, acc[1][1]);
    if (ldr) __builtin_amdgcn_s_wait_tensorcnt(0);
    __syncthreads();
  }

  const int hi = lane >> 4, nl = lane & 15;
#pragma unroll
  for (int ti = 0; ti < 2; ti++)
#pragma unroll
    for (int tj = 0; tj < 2; tj++) {
      int n = n0 + wn * 32 + tj * 16 + nl;
      float bs = bias[n];
      int t = n >> 8, cq = n & 255, head = cq >> 6, dd = cq & 63;
      __bf16* dst = (t == 0) ? qb : (t == 1) ? kb : vb;
      float sc = (t == 0) ? 0.125f : 1.0f;   // d^-0.5 folded into q
#pragma unroll
      for (int r = 0; r < 8; r++) {
        int m = m0 + wm * 32 + ti * 16 + r + 8 * hi;
        int bbb = m >> 12, p = m & (HW - 1);
        float val = (acc[ti][tj][r] + bs) * sc;
        dst[((size_t)(bbb * NH + head) * HW + p) * DH + dd] = (__bf16)val;
      }
    }
}

// ---------------------------------------------------------------------------
// Kernel 3: flash attention, one (b,head) x 128-q-row block per workgroup.
//   K tiles DMA'd by TDM (double-buffered); V manually transposed into LDS.
// ---------------------------------------------------------------------------
__global__ void __launch_bounds__(256) flash_attn_kernel(const __bf16* __restrict__ qb,
                                                         const __bf16* __restrict__ kb,
                                                         const __bf16* __restrict__ vb,
                                                         __bf16* __restrict__ ao) {
  __shared__ __align__(16) __bf16 Ks[2][64][72];   // [key][d], pitch 36 DW via TDM pad
  __shared__ __align__(16) __bf16 Vt[2][64][72];   // [d][key] (transposed)
  __shared__ __align__(16) __bf16 Ps[8][16][72];   // wave-private P scratch

  const int tid = threadIdx.x, lane = tid & 31, wv = tid >> 5;
  const int bh = blockIdx.y;
  const int q0 = blockIdx.x * 128;
  const int hi = lane >> 4, nl = lane & 15;
  const __bf16* kbase = kb + (size_t)bh * HW * DH;
  const __bf16* vbase = vb + (size_t)bh * HW * DH;

  // Q strip (16 x 64) -> 2 A-fragments straight from global (K-contiguous)
  const __bf16* qrow = qb + ((size_t)bh * HW + q0 + wv * 16) * DH;
  v16bf qf0 = load_frag(qrow,      DH, 0, lane);
  v16bf qf1 = load_frag(qrow + 32, DH, 0, lane);

  const v8f vzero = {0.f,0.f,0.f,0.f,0.f,0.f,0.f,0.f};
  v8f o[4] = {vzero, vzero, vzero, vzero};
  float mrow[8], lrow[8];
#pragma unroll
  for (int r = 0; r < 8; r++) { mrow[r] = -1e30f; lrow[r] = 0.f; }

  // prologue: stage tile 0
  if (tid == 0) tdm_load_2d(lds_off(&Ks[0][0][0]), kbase, 32, 64, 32, 4, 3);
  for (int i = tid; i < 512; i += 256) {
    int row = i >> 3, c8 = (i & 7) * 8;
    v8bf v = *(const v8bf*)(vbase + (size_t)row * DH + c8);
#pragma unroll
    for (int e = 0; e < 8; e++) Vt[0][c8 + e][row] = v[e];
  }
  if (tid == 0) __builtin_amdgcn_s_wait_tensorcnt(0);
  __syncthreads();

  for (int it = 0; it < HW / 64; ++it) {
    const int cur = it & 1, nxt = cur ^ 1;
    if (it + 1 < HW / 64) {  // stage next tile while computing current
      if (tid == 0)
        tdm_load_2d(lds_off(&Ks[nxt][0][0]), kbase + (size_t)(it + 1) * 64 * DH, 32, 64, 32, 4, 3);
      for (int i = tid; i < 512; i += 256) {
        int row = i >> 3, c8 = (i & 7) * 8;
        v8bf v = *(const v8bf*)(vbase + (size_t)(it + 1) * 64 * DH + row * DH + c8);
#pragma unroll
        for (int e = 0; e < 8; e++) Vt[nxt][c8 + e][row] = v[e];
      }
    }

    // --- S = Q K^T (16 x 64 per wave) ---
    v8f s[4];
#pragma unroll
    for (int jt = 0; jt < 4; jt++) {
      v16bf bk0 = load_frag(&Ks[cur][0][0],  72, jt * 16, lane);
      v16bf bk1 = load_frag(&Ks[cur][0][32], 72, jt * 16, lane);
      v8f z = vzero;
      z = wmma_bf16(qf0, bk0, z);
      z = wmma_bf16(qf1, bk1, z);
      s[jt] = z;
    }

    // --- online softmax (rows spread across 16-lane half-groups) ---
    float corr[8];
#pragma unroll
    for (int r = 0; r < 8; r++) {
      float mx = fmaxf(fmaxf(s[0][r], s[1][r]), fmaxf(s[2][r], s[3][r]));
      mx = rmax16(mx);
      float mnew = fmaxf(mrow[r], mx);
      float rsum = 0.f;
#pragma unroll
      for (int jt = 0; jt < 4; jt++) {
        float e = __expf(s[jt][r] - mnew);
        s[jt][r] = e;
        rsum += e;
      }
      rsum = rsum16(rsum);
      corr[r] = __expf(mrow[r] - mnew);
      lrow[r] = lrow[r] * corr[r] + rsum;
      mrow[r] = mnew;
    }
#pragma unroll
    for (int dt = 0; dt < 4; dt++)
#pragma unroll
      for (int r = 0; r < 8; r++) o[dt][r] *= corr[r];

    // --- relayout P: D-frag -> LDS -> A-frag (wave-private) ---
#pragma unroll
    for (int jt = 0; jt < 4; jt++)
#pragma unroll
      for (int r = 0; r < 8; r++)
        Ps[wv][r + 8 * hi][jt * 16 + nl] = (__bf16)s[jt][r];
    asm volatile("s_wait_dscnt 0" ::: "memory");
    v16bf pa0 = load_frag(&Ps[wv][0][0],  72, 0, lane);
    v16bf pa1 = load_frag(&Ps[wv][0][32], 72, 0, lane);

    // --- O += P V ---
#pragma unroll
    for (int dt = 0; dt < 4; dt++) {
      v16bf bv0 = load_frag(&Vt[cur][0][0],  72, dt * 16, lane);
      v16bf bv1 = load_frag(&Vt[cur][0][32], 72, dt * 16, lane);
      o[dt] = wmma_bf16(pa0, bv0, o[dt]);
      o[dt] = wmma_bf16(pa1, bv1, o[dt]);
    }
    if (tid == 0) __builtin_amdgcn_s_wait_tensorcnt(0);
    __syncthreads();
  }

  // --- normalize and store attn_out[b*hw+p, head*64+d] (row-major bf16) ---
  const int b = bh >> 2, head = bh & 3;
#pragma unroll
  for (int dt = 0; dt < 4; dt++)
#pragma unroll
    for (int r = 0; r < 8; r++) {
      int p = q0 + wv * 16 + r + 8 * hi;
      float val = o[dt][r] / lrow[r];
      ao[((size_t)b * HW + p) * CCH + head * DH + dt * 16 + nl] = (__bf16)val;
    }
}

// ---------------------------------------------------------------------------
// Kernel 4: proj GEMM + bias + residual, f32 output in [b,c,h,w]
// ---------------------------------------------------------------------------
__global__ void __launch_bounds__(256) proj_gemm_kernel(const __bf16* __restrict__ A,
                                                        const __bf16* __restrict__ W,
                                                        const float* __restrict__ bias,
                                                        const float* __restrict__ xres,
                                                        float* __restrict__ out) {
  __shared__ __align__(16) __bf16 As[2][128][40];
  __shared__ __align__(16) __bf16 Ws[2][64][40];
  const int tid = threadIdx.x, lane = tid & 31, wv = tid >> 5;
  const int wm = wv >> 1, wn = wv & 1;
  const int m0 = blockIdx.x * 128, n0 = blockIdx.y * 64;
  const bool ldr = (tid == 0);

  const v8f vzero = {0.f,0.f,0.f,0.f,0.f,0.f,0.f,0.f};
  v8f acc[2][2] = {{vzero, vzero}, {vzero, vzero}};

  if (ldr) {
    tdm_load_2d(lds_off(&As[0][0][0]), A + (size_t)m0 * CCH, 16, 128, 128, 3, 3);
    tdm_load_2d(lds_off(&Ws[0][0][0]), W + (size_t)n0 * CCH, 16,  64, 128, 3, 3);
    __builtin_amdgcn_s_wait_tensorcnt(0);
  }
  __syncthreads();

  for (int it = 0; it < 8; ++it) {
    const int cur = it & 1, nxt = cur ^ 1;
    if (it + 1 < 8 && ldr) {
      const int k0 = (it + 1) * 32;
      tdm_load_2d(lds_off(&As[nxt][0][0]), A + (size_t)m0 * CCH + k0, 16, 128, 128, 3, 3);
      tdm_load_2d(lds_off(&Ws[nxt][0][0]), W + (size_t)n0 * CCH + k0, 16,  64, 128, 3, 3);
    }
    v16bf a0 = load_frag(&As[cur][0][0], 40, wm * 32,      lane);
    v16bf a1 = load_frag(&As[cur][0][0], 40, wm * 32 + 16, lane);
    v16bf b0 = load_frag(&Ws[cur][0][0], 40, wn * 32,      lane);
    v16bf b1 = load_frag(&Ws[cur][0][0], 40, wn * 32 + 16, lane);
    acc[0][0] = wmma_bf16(a0, b0, acc[0][0]);
    acc[0][1] = wmma_bf16(a0, b1, acc[0][1]);
    acc[1][0] = wmma_bf16(a1, b0, acc[1][0]);
    acc[1][1] = wmma_bf16(a1, b1, acc[1][1]);
    if (ldr) __builtin_amdgcn_s_wait_tensorcnt(0);
    __syncthreads();
  }

  const int hi = lane >> 4, nl = lane & 15;
#pragma unroll
  for (int ti = 0; ti < 2; ti++)
#pragma unroll
    for (int tj = 0; tj < 2; tj++) {
      int n = n0 + wn * 32 + tj * 16 + nl;
      float bs = bias[n];
#pragma unroll
      for (int r = 0; r < 8; r++) {
        int m = m0 + wm * 32 + ti * 16 + r + 8 * hi;
        int bbb = m >> 12, p = m & (HW - 1);
        size_t oi = ((size_t)(bbb * CCH + n)) * HW + p;
        out[oi] = acc[ti][tj][r] + bs + xres[oi];
      }
    }
}

// ---------------------------------------------------------------------------
extern "C" void kernel_launch(void* const* d_in, const int* in_sizes, int n_in,
                              void* d_out, int out_size, void* d_ws, size_t ws_size,
                              hipStream_t stream) {
  const float* x      = (const float*)d_in[0];
  const float* gscale = (const float*)d_in[1];
  const float* gbias  = (const float*)d_in[2];
  const float* qkv_w  = (const float*)d_in[3];
  const float* qkv_b  = (const float*)d_in[4];
  const float* proj_w = (const float*)d_in[5];
  const float* proj_b = (const float*)d_in[6];
  float* out = (float*)d_out;

  char* ws = (char*)d_ws;
  const size_t MB8 = 8u << 20;
  __bf16* xn = (__bf16*)(ws);                          // [16384,256]   8 MiB
  __bf16* ao = (__bf16*)(ws + 1 * MB8);                // [16384,256]   8 MiB
  __bf16* qb = (__bf16*)(ws + 2 * MB8);                // [16,4096,64]  8 MiB
  __bf16* kb = (__bf16*)(ws + 3 * MB8);                // [16,4096,64]  8 MiB
  __bf16* vb = (__bf16*)(ws + 4 * MB8);                // [16,4096,64]  8 MiB
  __bf16* wq = (__bf16*)(ws + 5 * MB8);                // [768,256]  384 KiB
  __bf16* wp = (__bf16*)(ws + 5 * MB8 + (512u << 10)); // [256,256]  128 KiB

  cvt_bf16_kernel<<<768, 256, 0, stream>>>(qkv_w, wq, 768 * 256);
  cvt_bf16_kernel<<<256, 256, 0, stream>>>(proj_w, wp, 256 * 256);
  gn_kernel<<<128, 256, 0, stream>>>(x, gscale, gbias, xn);
  qkv_gemm_kernel<<<dim3(128, 12), 256, 0, stream>>>(xn, wq, qkv_b, qb, kb, vb);
  flash_attn_kernel<<<dim3(32, 16), 256, 0, stream>>>(qb, kb, vb, ao);
  proj_gemm_kernel<<<dim3(128, 4), 256, 0, stream>>>(ao, wp, proj_b, x, out);
}
